// naive_gat_like_55757265437281
// MI455X (gfx1250) — matrix operand order repaired and checked by reading, not compiled, not run on hardware
//
#include <hip/hip_runtime.h>
#include <math.h>

typedef __attribute__((ext_vector_type(16))) _Float16 v16h;
typedef __attribute__((ext_vector_type(8)))  float    v8f;
typedef __attribute__((ext_vector_type(2)))  float    v2f;

#define GN    8192
#define GDIM  512
#define GDK   128
#define GDV   128
#define NSPLT 2                    // split-K over j
#define JLEN  (GN / NSPLT)

#if __has_builtin(__builtin_amdgcn_global_load_async_to_lds_b64)
#define ATTN_ASYNC_STAGE 1
typedef __attribute__((ext_vector_type(2))) int v2i_t;
typedef __attribute__((address_space(1))) v2i_t* as1_v2i;
typedef __attribute__((address_space(3))) v2i_t* as3_v2i;
#endif

__device__ __forceinline__ unsigned f2key(float f) {
  unsigned u = __float_as_uint(f);
  return (u & 0x80000000u) ? ~u : (u | 0x80000000u);
}
__device__ __forceinline__ float key2f(unsigned k) {
  return (k & 0x80000000u) ? __uint_as_float(k ^ 0x80000000u)
                           : __uint_as_float(~k);
}

// ---------------------------------------------------------------------------
// Kernel 1: u1 = W^T a[:128], u2 = W^T a[128:], c1 = b.a1, c2 = b.a2,
//           init ordered-int max key for m_d.
// ---------------------------------------------------------------------------
__global__ void prep_kernel(const float* __restrict__ W, const float* __restrict__ b,
                            const float* __restrict__ a, float* __restrict__ u1,
                            float* __restrict__ u2, float* __restrict__ c12,
                            unsigned* __restrict__ mkey) {
  int bid = blockIdx.x;
  if (bid < 4) {
    int tid = bid * 256 + (int)threadIdx.x;       // 0..1023
    int d   = tid & 511;
    const float* av = (tid < 512) ? a : a + GDK;
    float acc = 0.f;
    #pragma unroll 4
    for (int k = 0; k < GDK; ++k) acc += W[k * GDIM + d] * av[k];
    if (tid < 512) u1[d] = acc; else u2[d] = acc;
  } else {
    __shared__ float r1[128], r2[128];
    int t = threadIdx.x;
    if (t < 128) { r1[t] = b[t] * a[t]; r2[t] = b[t] * a[GDK + t]; }
    __syncthreads();
    for (int s = 64; s > 0; s >>= 1) {
      if (t < s) { r1[t] += r1[t + s]; r2[t] += r2[t + s]; }
      __syncthreads();
    }
    if (t == 0) { c12[0] = r1[0]; c12[1] = r2[0]; *mkey = f2key(-INFINITY); }
  }
}

// ---------------------------------------------------------------------------
// Kernel 2: s_src[i] = x[i].u1 + c1 ; d_dst[i] = x[i].u2 + c2 ;
//           atomic ordered-int max over d_dst.  One wave32 per row.
// ---------------------------------------------------------------------------
__global__ void rowstats_kernel(const float* __restrict__ x, const float* __restrict__ u1,
                                const float* __restrict__ u2, const float* __restrict__ c12,
                                float* __restrict__ sSrc, float* __restrict__ dDst,
                                unsigned* __restrict__ mkey) {
  int wid = threadIdx.x >> 5, lane = threadIdx.x & 31;
  int row = blockIdx.x * 8 + wid;
  const float* xr = x + (size_t)row * GDIM;
  float a1 = 0.f, a2 = 0.f;
  #pragma unroll
  for (int kk = 0; kk < GDIM / 32; ++kk) {
    int idx = lane + kk * 32;
    float xv = xr[idx];
    a1 += xv * u1[idx];
    a2 += xv * u2[idx];
  }
  for (int off = 16; off; off >>= 1) {
    a1 += __shfl_xor(a1, off, 32);
    a2 += __shfl_xor(a2, off, 32);
  }
  if (lane == 0) {
    sSrc[row] = a1 + c12[0];
    float d = a2 + c12[1];
    dDst[row] = d;
    atomicMax(mkey, f2key(d));
  }
}

// ---------------------------------------------------------------------------
// Kernel 3: E[j] = exp(d_j - m_d) (<=1), C[j] = exp(0.01*d_j).
// ---------------------------------------------------------------------------
__global__ void expprep_kernel(const float* __restrict__ dDst,
                               const unsigned* __restrict__ mkey,
                               float* __restrict__ E, float* __restrict__ C) {
  int j = blockIdx.x * 256 + threadIdx.x;
  float md = key2f(*mkey);
  float d = dDst[j];
  E[j] = expf(d - md);
  C[j] = expf(0.01f * d);
}

// ---------------------------------------------------------------------------
// Kernel 4: v = x @ Wv^T + bv in exact f32 via v_wmma_f32_16x16x4_f32,
//           stored transposed + f16 as Vt[c][j]  (B-operand friendly).
// ---------------------------------------------------------------------------
__global__ void vproj_kernel(const float* __restrict__ x, const float* __restrict__ Wv,
                             const float* __restrict__ bv, _Float16* __restrict__ Vt) {
  int wid = threadIdx.x >> 5, lane = threadIdx.x & 31;
  int m0 = blockIdx.x * 16;          // x-row tile
  int n0 = wid * 16;                 // output-channel tile
  int rsel = lane & 15, hi = lane >> 4;

  v8f acc;
  #pragma unroll
  for (int k = 0; k < 8; ++k) acc[k] = 0.f;

  const float* xrow = x  + (size_t)(m0 + rsel) * GDIM + 2 * hi;
  const float* wrow = Wv + (size_t)(n0 + rsel) * GDIM + 2 * hi;
  #pragma unroll 8
  for (int k = 0; k < GDIM; k += 4) {
    v2f av = *(const v2f*)(xrow + k);
    v2f bw = *(const v2f*)(wrow + k);
    acc = __builtin_amdgcn_wmma_f32_16x16x4_f32(false, av, false, bw,
                                                (short)0, acc, false, false);
  }
  float bias = bv[n0 + rsel];
  _Float16* col = Vt + (size_t)(n0 + rsel) * GN + m0 + 8 * hi;
  #pragma unroll
  for (int k = 0; k < 8; ++k) col[k] = (_Float16)(acc[k] + bias);
}

// ---------------------------------------------------------------------------
// Kernel 5: fused softmax(P) @ V partials, flash-style.
// Grid = 64 row-blocks x NSPLT j-splits. Block = 512 thr (16 waves).
// Wave w: col-tile (w&7), row-half (w>>3) -> 4 row-tiles of 16.
// P tile 128x64 f16 generated in LDS from rank-1 factors (no exps in loop).
// Row sums via ones-B WMMA on owner waves. d/E/C staged through LDS with
// double-buffered gfx1250 async-to-LDS copies when available.
// ---------------------------------------------------------------------------
__global__ void __launch_bounds__(512) attn_kernel(
    const float* __restrict__ sSrc, const float* __restrict__ dDst,
    const float* __restrict__ E, const float* __restrict__ C,
    const _Float16* __restrict__ Vt, const unsigned* __restrict__ mkey,
    float* __restrict__ NumP, float* __restrict__ Zp) {
  __shared__ _Float16 Pt[128][72];         // padded rows: bank spread, 16B-aligned
  __shared__ float sS[128], sA[128], sB[128];
  __shared__ float sD[2][64], sE[2][64], sC[2][64];

  int rb = blockIdx.x >> 1;                // row-block 0..63
  int sp = blockIdx.x & 1;                 // j-split
  int r0 = rb * 128;
  int jbase = sp * JLEN;

  int t = threadIdx.x;
  int wid = t >> 5, lane = t & 31;
  int cwid = wid & 7, rh = wid >> 3;
  int colsel = lane & 15, hi = lane >> 4;
  int g  = t >> 6;                         // P-gen row group 0..7
  int jl = t & 63;                         // P-gen column within tile

  // ---- stage helpers: d/E/C tile (64 floats each) into LDS double buffer ---
  auto stage = [&](int nb, int jn) {
#ifdef ATTN_ASYNC_STAGE
    if (wid < 3) {
      const float* src = (wid == 0) ? (dDst + jn) : (wid == 1) ? (E + jn) : (C + jn);
      float* dst = (wid == 0) ? &sD[nb][0] : (wid == 1) ? &sE[nb][0] : &sC[nb][0];
      __builtin_amdgcn_global_load_async_to_lds_b64(
          (as1_v2i)(v2i_t*)(src + 2 * lane),
          (as3_v2i)(v2i_t*)(dst + 2 * lane), 0, 0);
    }
#else
    if (t < 64) {
      sD[nb][t] = dDst[jn + t];
      sE[nb][t] = E[jn + t];
      sC[nb][t] = C[jn + t];
    }
#endif
  };
  auto wait_stage = [&]() {
#ifdef ATTN_ASYNC_STAGE
#if __has_builtin(__builtin_amdgcn_s_wait_asynccnt)
    __builtin_amdgcn_s_wait_asynccnt(0);
#else
    asm volatile("s_wait_asynccnt 0x0" ::: "memory");
#endif
#endif
  };

  // ---- prologue: per-row softmax factors + first d/E/C stage -------------
  stage(0, jbase);
  float md = key2f(*mkey);
  if (t < 128) {
    float s = sSrc[r0 + t];
    float u = s + md;
    float M = (u > 0.f) ? u : 0.01f * u;   // closed-form row max of scores
    sS[t] = s;
    sA[t] = expf(u - M);                   // <= 1
    sB[t] = expf(0.01f * s - M);
  }
  wait_stage();
  __syncthreads();

  float myS[16], myA[16], myB[16];
  #pragma unroll
  for (int rr = 0; rr < 16; ++rr) {
    myS[rr] = sS[g * 16 + rr];
    myA[rr] = sA[g * 16 + rr];
    myB[rr] = sB[g * 16 + rr];
  }

  v8f acc[4];
  #pragma unroll
  for (int rt = 0; rt < 4; ++rt)
    #pragma unroll
    for (int k = 0; k < 8; ++k) acc[rt][k] = 0.f;
  v8f zacc;
  #pragma unroll
  for (int k = 0; k < 8; ++k) zacc[k] = 0.f;
  v16h ones;
  #pragma unroll
  for (int k = 0; k < 16; ++k) ones[k] = (_Float16)1.0f;

  int k0 = hi ? 16 : 0;
  const _Float16* bbase = Vt + (size_t)(cwid * 16 + colsel) * GN + jbase + hi * 16;

  for (int jt = 0; jt < JLEN; jt += 64) {
    int ib = (jt >> 6) & 1;
    // ---- generate 128x64 P tile (f16, A-operand-readable row-major) ----
    float d  = sD[ib][jl];
    float e  = sE[ib][jl];
    float cc = sC[ib][jl];
    #pragma unroll
    for (int rr = 0; rr < 16; ++rr) {
      float tt = myS[rr] + d;
      float p  = (tt > 0.f) ? myA[rr] * e : myB[rr] * cc;  // in [0,1]
      Pt[g * 16 + rr][jl] = (_Float16)p;
    }
    if (jt + 64 < JLEN) stage(ib ^ 1, jbase + jt + 64);
    __syncthreads();

    // ---- 2 K-chunks of 32, 4 row-tiles each: v_wmma_f32_16x16x32_f16 ----
    #pragma unroll
    for (int jc = 0; jc < 2; ++jc) {
      union { float4 f4[2]; v16h h; } B;
      const float4* gp = (const float4*)(bbase + jt + jc * 32);
      B.f4[0] = gp[0];
      B.f4[1] = gp[1];
      #pragma unroll
      for (int rtl = 0; rtl < 4; ++rtl) {
        int gt = rh * 4 + rtl;
        union { float4 f4[2]; v16h h; } A;
        const _Float16* prow = &Pt[gt * 16 + colsel][jc * 32 + k0];
        A.f4[0] = *(const float4*)prow;
        A.f4[1] = *(const float4*)(prow + 8);
        acc[rtl] = __builtin_amdgcn_wmma_f32_16x16x32_f16(
            false, A.h, false, B.h, (short)0, acc[rtl], false, false);
        if (cwid == rtl)   // owner wave: softmax denominator via ones-B WMMA
          zacc = __builtin_amdgcn_wmma_f32_16x16x32_f16(
              false, A.h, false, ones, (short)0, zacc, false, false);
      }
    }
    if (jt + 64 < JLEN) __builtin_prefetch(bbase + jt + 64, 0, 1);
    wait_stage();
    __syncthreads();
  }

  // ---- store partial Z rows (zacc replicated across N; take N=0 lanes) ----
  if (cwid < 4 && colsel == 0) {
    int gt = rh * 4 + cwid;
    #pragma unroll
    for (int k = 0; k < 8; ++k)
      Zp[(size_t)sp * GN + r0 + gt * 16 + 8 * hi + k] = zacc[k];
  }

  // ---- store partial numerators ----
  int outcol = cwid * 16 + colsel;
  float* nout = NumP + (size_t)sp * GN * GDV;
  #pragma unroll
  for (int rtl = 0; rtl < 4; ++rtl) {
    int gt = rh * 4 + rtl;
    #pragma unroll
    for (int k = 0; k < 8; ++k) {
      int row = r0 + gt * 16 + 8 * hi + k;
      nout[(size_t)row * GDV + outcol] = acc[rtl][k];
    }
  }
}

// ---------------------------------------------------------------------------
// Kernel 6: out = (Num0 + Num1) / (Z0 + Z1)
// ---------------------------------------------------------------------------
__global__ void reduce_kernel(const float* __restrict__ NumP,
                              const float* __restrict__ Zp,
                              float* __restrict__ out) {
  int idx = blockIdx.x * 256 + threadIdx.x;   // over GN*GDV
  int row = idx >> 7;
  float z = Zp[row] + Zp[GN + row];
  out[idx] = (NumP[idx] + NumP[(size_t)GN * GDV + idx]) / z;
}

// ---------------------------------------------------------------------------
extern "C" void kernel_launch(void* const* d_in, const int* in_sizes, int n_in,
                              void* d_out, int out_size, void* d_ws, size_t ws_size,
                              hipStream_t stream) {
  (void)in_sizes; (void)n_in; (void)out_size; (void)ws_size;
  const float* x  = (const float*)d_in[0];
  const float* W  = (const float*)d_in[1];
  const float* b  = (const float*)d_in[2];
  const float* Wv = (const float*)d_in[3];
  const float* bv = (const float*)d_in[4];
  const float* a  = (const float*)d_in[5];
  float* out = (float*)d_out;

  float* ws   = (float*)d_ws;
  float* u1   = ws;                        // 512
  float* u2   = ws + 512;                  // 512
  float* c12  = ws + 1024;                 // 2
  unsigned* mkey = (unsigned*)(ws + 1026);
  float* sSrc = ws + 2048;                 // 8192
  float* dDst = sSrc + GN;                 // 8192
  float* Earr = dDst + GN;                 // 8192
  float* Carr = Earr + GN;                 // 8192
  _Float16* Vt = (_Float16*)(Carr + GN);   // 128*8192 f16 = 2 MB (16B aligned)
  float* NumP = (float*)(Vt + (size_t)GDV * GN);   // NSPLT * 8192*128 f32 = 8 MB
  float* Zp   = NumP + (size_t)NSPLT * GN * GDV;   // NSPLT * 8192 f32

  prep_kernel    <<<5,              256, 0, stream>>>(W, b, a, u1, u2, c12, mkey);
  rowstats_kernel<<<GN / 8,         256, 0, stream>>>(x, u1, u2, c12, sSrc, dDst, mkey);
  expprep_kernel <<<GN / 256,       256, 0, stream>>>(dDst, mkey, Earr, Carr);
  vproj_kernel   <<<GN / 16,        256, 0, stream>>>(x, Wv, bv, Vt);
  attn_kernel    <<<(GN / 128) * NSPLT, 512, 0, stream>>>(sSrc, dDst, Earr, Carr,
                                                          Vt, mkey, NumP, Zp);
  reduce_kernel  <<<GN * GDV / 256, 256, 0, stream>>>(NumP, Zp, out);
}
